// Individual_80504866996713
// MI455X (gfx1250) — matrix operand order repaired and checked by reading, not compiled, not run on hardware
//
#include <hip/hip_runtime.h>
#include <hip/hip_bf16.h>

typedef __attribute__((ext_vector_type(16))) __bf16 v16bf;
typedef __attribute__((ext_vector_type(8)))  float  v8f;
typedef __attribute__((ext_vector_type(4)))  unsigned int v4u;
typedef __attribute__((ext_vector_type(8)))  int v8i;
typedef __attribute__((ext_vector_type(4)))  int v4i;

#define BATCH   512
#define FDIM    51200            // 200*256
#define KSPLIT  4                // K-chunks -> 4 partial Gram matrices
#define KCHUNK  (FDIM / KSPLIT)  // 12800
#define KTILE   64               // staged K per iteration (2 WMMA steps of K=32)
#define NIT     (KCHUNK / KTILE) // 200 iterations per workgroup
#define LDS_STRIDE 36            // u32 per LDS row = 144B (128B data + 16B TDM pad)
#define A_U32   (32 * LDS_STRIDE)
#define B_U32   (64 * LDS_STRIDE)
#define EPS 1e-8f

// round-to-nearest-even fp32 -> packed 2x bf16
__device__ __forceinline__ unsigned int pk_bf16(float a, float b) {
    unsigned int ua = __builtin_bit_cast(unsigned int, a);
    unsigned int ub = __builtin_bit_cast(unsigned int, b);
    ua += 0x7FFFu + ((ua >> 16) & 1u);
    ub += 0x7FFFu + ((ub >> 16) & 1u);
    return (ua >> 16) | (ub & 0xFFFF0000u);
}

// ---- TDM: load a (rows x 64) bf16 tile from a row-major [BATCH][FDIM] bf16
// tensor into LDS, hardware-padding each 128B row to a 144B stride. ----
__device__ __forceinline__ void tdm_load_tile(unsigned lds_byte_off,
                                              unsigned long long gaddr,
                                              int rows) {
    v4u g0;
    g0.x = 1u;                                            // count=1, user desc
    g0.y = lds_byte_off;                                  // lds_addr
    g0.z = (unsigned)(gaddr & 0xFFFFFFFFull);             // global_addr[31:0]
    g0.w = (unsigned)((gaddr >> 32) & 0x1FFFFFFull)       // global_addr[56:32]
         | (2u << 30);                                    // type=2 ("image")
    v8i g1;
    g1[0] = (1 << 16)       // data_size = 2 bytes
          | (1 << 20)       // pad_enable
          | (4 << 22)       // pad_interval: 32 DWORDs (=128B row)
          | (3 << 25);      // pad_amount:   4 DWORDs (=16B)
    g1[1] = (int)((FDIM & 0xFFFFu) << 16);                      // tensor_dim0 lo16
    g1[2] = (int)(((unsigned)FDIM >> 16) | ((unsigned)BATCH << 16)); // dim0 hi16 | dim1 lo16
    g1[3] = (int)(((unsigned)BATCH >> 16) | (64u << 16));       // dim1 hi16 | tile_dim0=64
    g1[4] = rows;                                               // tile_dim1 | tile_dim2=0
    g1[5] = (int)FDIM;                                          // tensor_dim0_stride lo32
    g1[6] = 0;                                                  // stride hi16 | dim1_stride lo16
    g1[7] = 0;
    v4i z4 = {0, 0, 0, 0};
    v8i z8 = {0, 0, 0, 0, 0, 0, 0, 0};
    // 6-arg variant (amdgpu-toolchain / clang-23): (g0, g1, g2, g3, g4, cpol)
    __builtin_amdgcn_tensor_load_to_lds(g0, g1, z4, z4, z8, 0);
}

// ---------------- Kernel 0: one-time fp32 -> bf16 (RNE) ----------------
__global__ void cvt_kernel(const float* __restrict__ f, unsigned int* __restrict__ ob) {
    const float2* src = (const float2*)f;
    const int n = FDIM * BATCH / 2;       // packed u32 count
    for (int i = blockIdx.x * blockDim.x + threadIdx.x; i < n;
         i += gridDim.x * blockDim.x) {
        float2 v = src[i];
        ob[i] = pk_bf16(v.x, v.y);
    }
}

// ---------------- Kernel 1: fp32 row norms ----------------
__global__ void norms_kernel(const float* __restrict__ f, float* __restrict__ norms) {
    __shared__ float red[256];
    const int row = blockIdx.x;
    const float4* p = (const float4*)(f + (size_t)row * FDIM);
    float s = 0.f;
    for (int k = threadIdx.x; k < FDIM / 4; k += 256) {
        float4 v = p[k];
        s = fmaf(v.x, v.x, s); s = fmaf(v.y, v.y, s);
        s = fmaf(v.z, v.z, s); s = fmaf(v.w, v.w, s);
    }
    red[threadIdx.x] = s; __syncthreads();
    for (int off = 128; off > 0; off >>= 1) {
        if (threadIdx.x < off) red[threadIdx.x] += red[threadIdx.x + off];
        __syncthreads();
    }
    if (threadIdx.x == 0) norms[row] = fmaxf(sqrtf(red[0]), EPS);
}

// ---- Kernel 2: partial Gram = F[:,kz] * F[:,kz]^T via double-buffered TDM + WMMA ----
// grid = (8, 16, 4): x -> 64-wide j tile, y -> 32-tall i tile, z -> K chunk
// block = 256 threads = 8 waves in a 2x4 arrangement of 16x16 WMMA tiles
__global__ void __launch_bounds__(256)
gram_kernel(const unsigned short* __restrict__ fb16, float* __restrict__ G) {
    __shared__ __align__(16) unsigned int lA[2][A_U32];
    __shared__ __align__(16) unsigned int lB[2][B_U32];

    const int tid  = threadIdx.x;
    const int lane = tid & 31;
    const int wave = tid >> 5;
    const int wm   = wave & 1;       // 0..1 : 16-row slice of 32-row A panel
    const int wn   = wave >> 1;      // 0..3 : 16-col slice of 64-col B panel
    const int ibase = blockIdx.y * 32;
    const int jbase = blockIdx.x * 64;

    // LDS aperture maps flat addr[31:0] directly to the LDS byte offset (ISA 10.2)
    const unsigned ldsA0 = (unsigned)(unsigned long long)(void*)&lA[0][0];
    const unsigned ldsA1 = (unsigned)(unsigned long long)(void*)&lA[1][0];
    const unsigned ldsB0 = (unsigned)(unsigned long long)(void*)&lB[0][0];
    const unsigned ldsB1 = (unsigned)(unsigned long long)(void*)&lB[1][0];
    // byte address of this chunk's K origin within each panel's first row
    const unsigned long long kz = (unsigned long long)blockIdx.z * KCHUNK * 2ull;
    const unsigned long long gA =
        (unsigned long long)(void*)(fb16 + (size_t)ibase * FDIM) + kz;
    const unsigned long long gB =
        (unsigned long long)(void*)(fb16 + (size_t)jbase * FDIM) + kz;

    // two independent accumulators (one per K=32 sub-step) -> no WMMA->WMMA
    // D->C register dependency inside an iteration, no hazard NOPs / mov chains
    v8f acc0 = {};
    v8f acc1 = {};

    // ---- software pipeline prologue: tiles 0 and 1 in flight ----
    if (wave == 0) {
        tdm_load_tile(ldsA0, gA, 32);
        tdm_load_tile(ldsB0, gB, 64);
        tdm_load_tile(ldsA1, gA + KTILE * 2, 32);
        tdm_load_tile(ldsB1, gB + KTILE * 2, 64);
    }

    for (int it = 0; it < NIT; ++it) {
        const int cur = it & 1;
        if (wave == 0) {
            // TDM ops complete in order: <=2 outstanding means tile `it` landed
            if (it + 1 < NIT) __builtin_amdgcn_s_wait_tensorcnt(2);
            else              __builtin_amdgcn_s_wait_tensorcnt(0);
        }
        __syncthreads();   // publish buffer `cur` to all waves

        // ---- two independent WMMA steps of K=32 from buffer `cur` ----
        const unsigned int* rA = &lA[cur][(wm * 16 + (lane & 15)) * LDS_STRIDE];
        const unsigned int* rB = &lB[cur][(wn * 16 + (lane & 15)) * LDS_STRIDE];
        const int g = lane >> 4;
        union { uint4 q[2]; v16bf v; } a0, b0, a1, b1;
        // A fragment: lane=(g, m=lane&15); VGPR j holds
        // K = s*32 + (j>>2)*16 + g*8 + (j&3)*2 (+1) -> two 16B chunks
        a0.q[0] = *(const uint4*)&rA[g * 4];
        a0.q[1] = *(const uint4*)&rA[8 + g * 4];
        a1.q[0] = *(const uint4*)&rA[16 + g * 4];
        a1.q[1] = *(const uint4*)&rA[16 + 8 + g * 4];
        // B fragment: lane=(h=g, n=lane&15); column n, same K layout as A
        b0.q[0] = *(const uint4*)&rB[g * 4];
        b0.q[1] = *(const uint4*)&rB[8 + g * 4];
        b1.q[0] = *(const uint4*)&rB[16 + g * 4];
        b1.q[1] = *(const uint4*)&rB[16 + 8 + g * 4];

        acc0 = __builtin_amdgcn_wmma_f32_16x16x32_bf16(
            false, a0.v, false, b0.v, (short)0, acc0, false, false);
        acc1 = __builtin_amdgcn_wmma_f32_16x16x32_bf16(
            false, a1.v, false, b1.v, (short)0, acc1, false, false);

        __syncthreads();   // release buffer `cur` before tile it+2 overwrites it

        if (wave == 0 && it + 2 < NIT) {
            const unsigned long long koff = (unsigned long long)(it + 2) * (KTILE * 2);
            tdm_load_tile(cur ? ldsA1 : ldsA0, gA + koff, 32);
            tdm_load_tile(cur ? ldsB1 : ldsB0, gB + koff, 64);
        }
    }

    // ---- store 16x16 fp32 tile: VGPR r, lane L -> M = r + (L>>4)*8, N = L&15 ----
    const int cn = lane & 15, ch = lane >> 4;
    float* out = G + (size_t)blockIdx.z * (BATCH * BATCH)
                   + (size_t)(ibase + wm * 16) * BATCH + jbase + wn * 16;
#pragma unroll
    for (int r = 0; r < 8; ++r)
        out[(size_t)(ch * 8 + r) * BATCH + cn] = acc0[r] + acc1[r];
}

// ---------------- Kernel 3: contrastive loss over summed partials ----------------
__global__ void loss_kernel(const float* __restrict__ G, const float* __restrict__ norms,
                            const float* __restrict__ labels, float* __restrict__ out) {
    __shared__ float nrm[BATCH];
    __shared__ unsigned char lab[BATCH];
    __shared__ float red[1024];
    const int t = threadIdx.x;
    if (t < BATCH) {
        nrm[t] = norms[t];
        lab[t] = (labels[2 * t + 1] > labels[2 * t]) ? 1 : 0;   // argmax, first-on-tie
    }
    __syncthreads();
    const int S = BATCH * BATCH;
    float s = 0.f;
    for (int idx = t; idx < S; idx += 1024) {
        const int i = idx >> 9, j = idx & (BATCH - 1);
        if (i == j) continue;
        const float dot = (G[idx] + G[idx + S]) + (G[idx + 2 * S] + G[idx + 3 * S]);
        const float sim  = dot / (nrm[i] * nrm[j]);
        const float term = (lab[i] == lab[j]) ? (1.f - sim) : fmaxf(0.f, 1.f - sim);
        s += term;
    }
    red[t] = s; __syncthreads();
    for (int off = 512; off > 0; off >>= 1) {
        if (t < off) red[t] += red[t + off];
        __syncthreads();
    }
    if (t == 0) out[0] = red[0] / (float)(BATCH * (BATCH - 1));
}

extern "C" void kernel_launch(void* const* d_in, const int* in_sizes, int n_in,
                              void* d_out, int out_size, void* d_ws, size_t ws_size,
                              hipStream_t stream) {
    const float* feat   = (const float*)d_in[0];   // [512, 200, 256] fp32
    const float* labels = (const float*)d_in[1];   // [512, 2] fp32
    float* out   = (float*)d_out;                  // scalar fp32

    float*        norms = (float*)d_ws;                         // 512 f32
    float*        G     = (float*)d_ws + BATCH;                 // 4 * 512*512 f32
    unsigned int* fb16  = (unsigned int*)((float*)d_ws + BATCH
                                          + KSPLIT * BATCH * BATCH); // 52.4MB bf16

    cvt_kernel<<<2048, 256, 0, stream>>>(feat, fb16);
    norms_kernel<<<BATCH, 256, 0, stream>>>(feat, norms);
    dim3 grid(BATCH / 64, BATCH / 32, KSPLIT);      // (8, 16, 4) = 512 workgroups
    gram_kernel<<<grid, 256, 0, stream>>>((const unsigned short*)fb16, G);
    loss_kernel<<<1, 1024, 0, stream>>>(G, norms, labels, out);
}
